// Attention_74861279969698
// MI455X (gfx1250) — matrix operand order repaired and checked by reading, not compiled, not run on hardware
//
#include <hip/hip_runtime.h>
#include <cstdint>

typedef __bf16 bf16_t;
typedef __attribute__((ext_vector_type(16))) __bf16 v16bf;
typedef __attribute__((ext_vector_type(8)))  __bf16 v8bf;
typedef __attribute__((ext_vector_type(8)))  float  v8f;

#define WMMA_BF16(a, b, c) \
  __builtin_amdgcn_wmma_f32_16x16x32_bf16(false, (a), false, (b), (short)0, (c), false, false)

static __device__ __forceinline__ v16bf frag2(const bf16_t* p0, const bf16_t* p1) {
  v16bf r;
  *((v8bf*)&r)       = *((const v8bf*)p0);
  *(((v8bf*)&r) + 1) = *((const v8bf*)p1);
  return r;
}

// Async copy of 16 bytes global -> LDS (ASYNCcnt-tracked, no VGPR data path).
static __device__ __forceinline__ void async_b128(const void* gsrc, void* lds_dst) {
  uint32_t la = (uint32_t)(uintptr_t)lds_dst;  // low 32 bits = wave LDS offset
  asm volatile("global_load_async_to_lds_b128 %0, %1, off"
               :: "v"(la), "v"(gsrc) : "memory");
}
static __device__ __forceinline__ void wait_async0() {
  asm volatile("s_wait_asynccnt 0" ::: "memory");
}

// ---------------------------------------------------------------------------
// fp32 -> bf16 conversion (grid-stride)
// ---------------------------------------------------------------------------
__global__ void cvt_f32_bf16(const float* __restrict__ s, bf16_t* __restrict__ d, long n) {
  long i  = (long)blockIdx.x * blockDim.x + threadIdx.x;
  long st = (long)gridDim.x * blockDim.x;
  for (; i < n; i += st) d[i] = (bf16_t)s[i];
}

// ---------------------------------------------------------------------------
// WMMA GEMM:  out[M,Nout] = A[M,K] * W[Nout,K]^T  (+bias epilogues)
// Double-buffered async (GLOBAL_LOAD_ASYNC_TO_LDS_B128) tile pipeline.
// MODE 0: QKV — scatter bf16 into q/k/v [B,H,N,Dh]; q gets (x@Wq + q_bias)*scale
// MODE 1: proj — fp32 out = acc + proj_bias
// Block tile 128x128x32, 8 waves (4x2), wave tile 32x64 (2x4 WMMA tiles)
// ---------------------------------------------------------------------------
#define BM 128
#define BN 128
#define BK 32
#define LDT 40  // padded LDS row stride (halfwords): conflict-free b128 loads

template <int MODE>
__global__ __launch_bounds__(256) void gemm_wmma(
    const bf16_t* __restrict__ A, const bf16_t* __restrict__ W,
    const float* __restrict__ bias0, const float* __restrict__ bias1,
    bf16_t* __restrict__ qo, bf16_t* __restrict__ ko, bf16_t* __restrict__ vo,
    float* __restrict__ fout, int M, int Nout, int K, int Nseq) {
  __shared__ __align__(16) bf16_t As[2][BM * LDT];
  __shared__ __align__(16) bf16_t Bs[2][BN * LDT];

  const int tid  = threadIdx.x;
  const int lane = tid & 31;
  const int wave = tid >> 5;
  const int wm   = wave >> 1;  // 0..3 (M)
  const int wn   = wave & 1;   // 0..1 (N)
  const int blockM = blockIdx.y * BM;
  const int blockN = blockIdx.x * BN;

  const int rl   = lane & 15;
  const int kc8  = (lane >> 4) * 8;   // A-frag K chunk base
  const int kc16 = (lane >> 4) * 16;  // unused; B-frag also uses kc8

  v8f acc[2][4] = {};

  const int lrow   = tid >> 2;        // 0..63
  const int lchunk = (tid & 3) * 8;   // 0,8,16,24

  // issue the two 16B async fills this thread owns per tile buffer
  auto issue_tile = [&](int buf, int k0) {
#pragma unroll
    for (int p = 0; p < 2; ++p) {
      int row = lrow + p * 64;
      async_b128(A + (size_t)(blockM + row) * K + k0 + lchunk,
                 &As[buf][row * LDT + lchunk]);
      async_b128(W + (size_t)(blockN + row) * K + k0 + lchunk,
                 &Bs[buf][row * LDT + lchunk]);
    }
  };

  issue_tile(0, 0);

  for (int k0 = 0; k0 < K; k0 += BK) {
    const int buf = (k0 / BK) & 1;
    wait_async0();    // this wave's fills for `buf` are in LDS
    __syncthreads();  // everyone's fills done AND prev buffer's compute done
    if (k0 + BK < K) issue_tile(buf ^ 1, k0 + BK);

    v16bf af[2], bfg[4];
#pragma unroll
    for (int mt = 0; mt < 2; ++mt) {
      const bf16_t* p = &As[buf][(wm * 32 + mt * 16 + rl) * LDT];
      af[mt] = frag2(p + kc8, p + kc8 + 16);
    }
#pragma unroll
    for (int nt = 0; nt < 4; ++nt) {
      const bf16_t* p = &Bs[buf][(wn * 64 + nt * 16 + rl) * LDT];
      bfg[nt] = frag2(p + kc8, p + kc8 + 16);
    }
#pragma unroll
    for (int mt = 0; mt < 2; ++mt)
#pragma unroll
      for (int nt = 0; nt < 4; ++nt)
        acc[mt][nt] = WMMA_BF16(af[mt], bfg[nt], acc[mt][nt]);
  }

  // -------- epilogue --------
#pragma unroll
  for (int mt = 0; mt < 2; ++mt) {
#pragma unroll
    for (int nt = 0; nt < 4; ++nt) {
      const int j0 = blockN + wn * 64 + nt * 16;
      const int jl = j0 + rl;
      if (MODE == 0) {
        const int Cq  = Nout / 3;                    // 1024
        const int sel = j0 / Cq;                     // 0=q 1=k 2=v (tile-uniform)
        const int cc  = jl - sel * Cq;
        const int h   = cc >> 6;
        const int d   = cc & 63;
        const float bias = (sel == 0) ? bias0[cc] : (sel == 2) ? bias1[cc] : 0.0f;
        const float mul  = (sel == 0) ? 0.125f : 1.0f;  // Dh^-0.5 = 64^-0.5
        bf16_t* dst = (sel == 0) ? qo : (sel == 1) ? ko : vo;
#pragma unroll
        for (int r = 0; r < 8; ++r) {
          int m = blockM + wm * 32 + mt * 16 + r + ((lane >> 4) << 3);
          int b = m / Nseq;
          int n = m - b * Nseq;
          float val = (acc[mt][nt][r] + bias) * mul;
          dst[((size_t)(b * 16 + h) * Nseq + n) * 64 + d] = (bf16_t)val;
        }
      } else {
        const float bias = bias0[jl];
#pragma unroll
        for (int r = 0; r < 8; ++r) {
          int m = blockM + wm * 32 + mt * 16 + r + ((lane >> 4) << 3);
          fout[(size_t)m * Nout + jl] = acc[mt][nt][r] + bias;
        }
      }
    }
  }
}

// ---------------------------------------------------------------------------
// Flash attention: block = (b,h, 64 q-rows), 4 waves x 16 rows each.
// Online softmax over k-tiles of 64, rel_pos_bias (NT loads) + mask fused.
// q pre-scaled. out written bf16 as [B,N,H*Dh].
// ---------------------------------------------------------------------------
__global__ __launch_bounds__(128) void attn_wmma(
    const bf16_t* __restrict__ q, const bf16_t* __restrict__ kk,
    const bf16_t* __restrict__ v, const float* __restrict__ bias,
    const unsigned char* __restrict__ mask, bf16_t* __restrict__ out, int Nseq) {
  __shared__ __align__(16) bf16_t vT[64 * 72];       // V^T tile [Dh][kc]
  __shared__ __align__(16) bf16_t Pls[4 * 16 * 72];  // per-wave P staging

  const int tid  = threadIdx.x;
  const int lane = tid & 31;
  const int wave = tid >> 5;
  const int bh = blockIdx.y;  // b*16 + h
  const int b = bh >> 4, h = bh & 15;
  const int m0 = blockIdx.x * 64 + wave * 16;
  const size_t headbase = (size_t)bh * Nseq * 64;

  const int rl   = lane & 15;
  const int kc8  = (lane >> 4) * 8;
  const int kc16 = (lane >> 4) * 16;
  const int mhi  = (lane >> 4) << 3;  // +8 row offset for upper half-wave

  // Q A-fragments (K = Dh = 64 -> two frags), contiguous rows in [B,H,N,Dh]
  const bf16_t* qrow = q + headbase + (size_t)(m0 + rl) * 64;
  const v16bf aq0 = frag2(qrow + kc8, qrow + kc8 + 16);
  const v16bf aq1 = frag2(qrow + 32 + kc8, qrow + 32 + kc8 + 16);

  v8f O[4] = {};
  float rmax[8], rsum[8];
#pragma unroll
  for (int r = 0; r < 8; ++r) { rmax[r] = -1e30f; rsum[r] = 0.0f; }

  for (int kc0 = 0; kc0 < Nseq; kc0 += 64) {
    __syncthreads();  // all waves done reading previous vT
    // cooperative transpose of V tile [64 kcols x 64 Dh] into vT[Dh][kc]
#pragma unroll
    for (int p = 0; p < 4; ++p) {
      int chunk = tid + p * 128;      // 0..511
      int kr    = chunk >> 3;         // k-row 0..63
      int dblk  = (chunk & 7) * 8;    // Dh chunk
      v8bf vv = *(const v8bf*)(v + headbase + (size_t)(kc0 + kr) * 64 + dblk);
#pragma unroll
      for (int i = 0; i < 8; ++i) vT[(dblk + i) * 72 + kr] = vv[i];
    }
    __syncthreads();

    // S = q @ k^T for this wave's 16 rows x 64 k-cols, + bias + mask
    v8f S[4];
#pragma unroll
    for (int nt = 0; nt < 4; ++nt) {
      const int kcol = kc0 + nt * 16 + rl;  // B-matrix column = k row (contiguous)
      const bf16_t* krow = kk + headbase + (size_t)kcol * 64;
      v16bf bk0 = frag2(krow + kc8, krow + kc8 + 16);
      v16bf bk1 = frag2(krow + 32 + kc8, krow + 32 + kc8 + 16);
      v8f s = {};
      s = WMMA_BF16(aq0, bk0, s);
      s = WMMA_BF16(aq1, bk1, s);
      const bool mk = mask[b * Nseq + kcol] != 0;
#pragma unroll
      for (int r = 0; r < 8; ++r) {
        int m = m0 + r + mhi;
        float bv = __builtin_nontemporal_load(&bias[((size_t)h * Nseq + m) * Nseq + kcol]);
        float sv = s[r] + bv;
        s[r] = mk ? sv : -1e30f;
      }
      S[nt] = s;
    }

    // online softmax: rows live in 16-lane half-waves (xor<16 stays in half)
#pragma unroll
    for (int r = 0; r < 8; ++r) {
      float tmax = fmaxf(fmaxf(S[0][r], S[1][r]), fmaxf(S[2][r], S[3][r]));
#pragma unroll
      for (int off = 1; off < 16; off <<= 1) tmax = fmaxf(tmax, __shfl_xor(tmax, off, 32));
      float nm   = fmaxf(rmax[r], tmax);
      float corr = __expf(rmax[r] - nm);
      float psum = 0.0f;
#pragma unroll
      for (int nt = 0; nt < 4; ++nt) {
        float p = __expf(S[nt][r] - nm);
        S[nt][r] = p;
        psum += p;
      }
#pragma unroll
      for (int off = 1; off < 16; off <<= 1) psum += __shfl_xor(psum, off, 32);
      rsum[r] = rsum[r] * corr + psum;
      rmax[r] = nm;
#pragma unroll
      for (int dt = 0; dt < 4; ++dt) O[dt][r] *= corr;
    }

    // stage P (C-layout -> A-layout) through this wave's private LDS region
    bf16_t* pw = &Pls[wave * 16 * 72];
#pragma unroll
    for (int nt = 0; nt < 4; ++nt)
#pragma unroll
      for (int r = 0; r < 8; ++r)
        pw[(r + mhi) * 72 + nt * 16 + rl] = (bf16_t)S[nt][r];
    asm volatile("s_wait_dscnt 0" ::: "memory");  // DS in-order, wave-private

    // O += P @ V   (K = 64 k-cols -> 2 k-steps; N = Dh)
#pragma unroll
    for (int ks = 0; ks < 2; ++ks) {
      v16bf ap = frag2(&pw[rl * 72 + ks * 32 + kc8], &pw[rl * 72 + ks * 32 + kc8 + 16]);
#pragma unroll
      for (int dt = 0; dt < 4; ++dt) {
        const bf16_t* vp = &vT[(dt * 16 + rl) * 72 + ks * 32];
        v16bf bv = frag2(vp + kc8, vp + kc8 + 16);
        O[dt] = WMMA_BF16(ap, bv, O[dt]);
      }
    }
  }

  // normalize + write bf16 [B,N,C] with c = h*64 + d
#pragma unroll
  for (int r = 0; r < 8; ++r) {
    float inv = 1.0f / rsum[r];
    int m = m0 + r + mhi;
#pragma unroll
    for (int dt = 0; dt < 4; ++dt) {
      int c = h * 64 + dt * 16 + rl;
      out[((size_t)b * Nseq + m) * 1024 + c] = (bf16_t)(O[dt][r] * inv);
    }
  }
}

// ---------------------------------------------------------------------------
extern "C" void kernel_launch(void* const* d_in, const int* in_sizes, int n_in,
                              void* d_out, int out_size, void* d_ws, size_t ws_size,
                              hipStream_t stream) {
  const int B = 4, N = 2048, C = 1024, H = 16;
  const int M = B * N;  // 8192

  const float* x         = (const float*)d_in[0];
  const float* qkv_w     = (const float*)d_in[1];
  const float* q_bias    = (const float*)d_in[2];
  const float* v_bias    = (const float*)d_in[3];
  const float* proj_w    = (const float*)d_in[4];
  const float* proj_bias = (const float*)d_in[5];
  const float* rel       = (const float*)d_in[6];
  const unsigned char* mask = (const unsigned char*)d_in[7];
  float* out = (float*)d_out;

  char* ws = (char*)d_ws;
  size_t off = 0;
  auto alloc = [&](size_t bytes) -> void* {
    void* p = ws + off;
    off += (bytes + 255) & ~(size_t)255;
    return p;
  };
  bf16_t* xb    = (bf16_t*)alloc((size_t)M * C * 2);          // 16 MB
  bf16_t* wqkv  = (bf16_t*)alloc((size_t)3 * C * C * 2);      //  6 MB
  bf16_t* wproj = (bf16_t*)alloc((size_t)C * C * 2);          //  2 MB
  bf16_t* qb    = (bf16_t*)alloc((size_t)M * C * 2);          // 16 MB [B,H,N,64]
  bf16_t* kb    = (bf16_t*)alloc((size_t)M * C * 2);          // 16 MB
  bf16_t* vb    = (bf16_t*)alloc((size_t)M * C * 2);          // 16 MB
  bf16_t* ao    = (bf16_t*)alloc((size_t)M * C * 2);          // 16 MB [B,N,C]
  (void)ws_size; (void)in_sizes; (void)n_in; (void)out_size;

  cvt_f32_bf16<<<2048, 256, 0, stream>>>(x, xb, (long)M * C);
  cvt_f32_bf16<<<1024, 256, 0, stream>>>(qkv_w, wqkv, (long)3 * C * C);
  cvt_f32_bf16<<<512, 256, 0, stream>>>(proj_w, wproj, (long)C * C);

  // QKV: [8192,3072] = xb[8192,1024] @ wqkv[3072,1024]^T
  gemm_wmma<0><<<dim3(3 * C / BN, M / BM), 256, 0, stream>>>(
      xb, wqkv, q_bias, v_bias, qb, kb, vb, nullptr, M, 3 * C, C, N);

  // attention: grid = (q-tiles of 64, B*H)
  attn_wmma<<<dim3(N / 64, B * H), 128, 0, stream>>>(qb, kb, vb, rel, mask, ao, N);

  // proj: out[8192,1024] = ao @ wproj^T + proj_bias
  gemm_wmma<1><<<dim3(C / BN, M / BM), 256, 0, stream>>>(
      ao, wproj, proj_bias, nullptr, nullptr, nullptr, nullptr, out, M, C, C, N);
}
